// Non_local_MH_31799937859997
// MI455X (gfx1250) — matrix operand order repaired and checked by reading, not compiled, not run on hardware
//
#include <hip/hip_runtime.h>
#include <stdint.h>

#define CCH   512
#define NSP   1024
#define NB    8          // batch
#define EPSBN 1e-5f

typedef __attribute__((ext_vector_type(16))) __bf16 v16bf;
typedef __attribute__((ext_vector_type(8)))  float  v8f;
typedef __attribute__((ext_vector_type(4)))  unsigned int v4u;

static __device__ __forceinline__ unsigned short f2bf(float x) {
    unsigned int b = __float_as_uint(x);
    b += 0x7FFFu + ((b >> 16) & 1u);       // round-to-nearest-even
    return (unsigned short)(b >> 16);
}

static __device__ __forceinline__ v8f vzero8() {
    v8f z = {0.f,0.f,0.f,0.f,0.f,0.f,0.f,0.f};
    return z;
}

// A-operand fragment (16x32 bf16) from a row-major [M][K] source, stride ldk (elems).
// Lane (half = lane/16, r = lane%16): row = row0 + r, K = k0 + 16*half + 0..15 (contiguous).
static __device__ __forceinline__ v16bf load_a_frag(const unsigned short* base, int row0, int k0, int ldk) {
    int lane = (int)(threadIdx.x & 31u);
    int half = lane >> 4, r = lane & 15;
    const unsigned short* p = base + (size_t)(row0 + r) * ldk + k0 + half * 16;
    union { v16bf bf; v4u u[2]; } f;
    f.u[0] = *(const v4u*)(p);
    f.u[1] = *(const v4u*)(p + 8);
    return f.bf;
}

// B-operand fragment (32x16 bf16) from a "B^T" row-major [N][K] source, stride ldk.
// Lane: N = n0 + r, K = k0 + 16*half + 0..15 (contiguous).
static __device__ __forceinline__ v16bf load_b_frag(const unsigned short* base, int n0, int k0, int ldk) {
    int lane = (int)(threadIdx.x & 31u);
    int half = lane >> 4, r = lane & 15;
    const unsigned short* p = base + (size_t)(n0 + r) * ldk + k0 + half * 16;
    union { v16bf bf; v4u u[2]; } f;
    f.u[0] = *(const v4u*)(p);
    f.u[1] = *(const v4u*)(p + 8);
    return f.bf;
}

static __device__ __forceinline__ v8f wmma_bf16(v16bf a, v16bf b, v8f c) {
    return __builtin_amdgcn_wmma_f32_16x16x32_bf16(false, a, false, b, (short)0, c, false, false);
}

// ---------------- conversion kernels ----------------

__global__ __launch_bounds__(256) void k_cvt_w(const float* __restrict__ w0, const float* __restrict__ w1,
                                               const float* __restrict__ w2, const float* __restrict__ w3,
                                               unsigned short* __restrict__ dst) {
    int idx = blockIdx.x * 256 + threadIdx.x;            // 4 * 262144 total
    int which = idx >> 18;
    int off = idx & 262143;
    const float* s = (which == 0) ? w0 : ((which == 1) ? w1 : ((which == 2) ? w2 : w3));
    dst[idx] = f2bf(s[off]);
}

// X [b][c][n] f32 -> XT [b][n][c] bf16
__global__ __launch_bounds__(256) void k_cvt_x(const float* __restrict__ X, unsigned short* __restrict__ XT) {
    int idx = blockIdx.x * 256 + threadIdx.x;            // 8*512*1024
    int n = idx & (NSP - 1);
    int c = (idx >> 10) & (CCH - 1);
    int b = idx >> 19;
    XT[(size_t)b * NSP * CCH + (size_t)n * CCH + c] = f2bf(X[idx]);
}

__global__ __launch_bounds__(256) void k_zero(float* __restrict__ p, int n) {
    int i = blockIdx.x * 256 + threadIdx.x;
    if (i < n) p[i] = 0.f;
}

// ---------------- GEMM 1: fused QKV convs ----------------
// mat 0 (Wq) -> "K" stored [b][c][n]; mat 1 (Wk) -> "Q" stored transposed [b][n][c];
// mat 2 (Wv) -> "V" stored [b][c][n]. (Reference swaps names exactly this way.)
__global__ __launch_bounds__(256) void k_gemm_qkv(const unsigned short* __restrict__ Wbf,  // [3][512][512]
                                                  const unsigned short* __restrict__ XT,   // [B][n][c]
                                                  const float* __restrict__ bq,
                                                  const float* __restrict__ bk,
                                                  const float* __restrict__ bv,
                                                  unsigned short* __restrict__ Kbuf,
                                                  unsigned short* __restrict__ QT,
                                                  unsigned short* __restrict__ Vbuf) {
    int nt = blockIdx.x, mt = blockIdx.y;
    int b = blockIdx.z / 3, mat = blockIdx.z % 3;
    int m0 = mt * 64, n0 = nt * 128;
    int w = (int)(threadIdx.x >> 5);
    int lane = (int)(threadIdx.x & 31u);
    int half = lane >> 4, lr = lane & 15;
    int mw = m0 + (w & 1) * 32;
    int nw = n0 + (w >> 1) * 32;
    const unsigned short* A  = Wbf + (size_t)mat * CCH * CCH;
    const unsigned short* Bt = XT + (size_t)b * NSP * CCH;

    v8f acc[2][2];
    acc[0][0] = vzero8(); acc[0][1] = vzero8(); acc[1][0] = vzero8(); acc[1][1] = vzero8();

    for (int k0 = 0; k0 < CCH; k0 += 32) {
        v16bf a0 = load_a_frag(A, mw,      k0, CCH);
        v16bf a1 = load_a_frag(A, mw + 16, k0, CCH);
        v16bf b0 = load_b_frag(Bt, nw,      k0, CCH);
        v16bf b1 = load_b_frag(Bt, nw + 16, k0, CCH);
        acc[0][0] = wmma_bf16(a0, b0, acc[0][0]);
        acc[0][1] = wmma_bf16(a0, b1, acc[0][1]);
        acc[1][0] = wmma_bf16(a1, b0, acc[1][0]);
        acc[1][1] = wmma_bf16(a1, b1, acc[1][1]);
    }

    const float* bias = (mat == 0) ? bq : ((mat == 1) ? bk : bv);

    if (mat == 1) {           // transposed store: Q^T [n][c], one b128 per tile per lane
        unsigned short* dst = QT + (size_t)b * NSP * CCH;
#pragma unroll
        for (int i = 0; i < 2; i++)
#pragma unroll
        for (int j = 0; j < 2; j++) {
            int n  = nw + j * 16 + lr;
            int cb = mw + i * 16 + half * 8;
            union { v4u v; unsigned int u[4]; } pk;
#pragma unroll
            for (int r = 0; r < 4; r++) {
                unsigned lo = f2bf(acc[i][j][2 * r]     + bias[cb + 2 * r]);
                unsigned hi = f2bf(acc[i][j][2 * r + 1] + bias[cb + 2 * r + 1]);
                pk.u[r] = lo | (hi << 16);
            }
            *(v4u*)(dst + (size_t)n * CCH + cb) = pk.v;
        }
    } else {                  // natural store [c][n] for K / V
        unsigned short* dst = ((mat == 0) ? Kbuf : Vbuf) + (size_t)b * CCH * NSP;
#pragma unroll
        for (int i = 0; i < 2; i++)
#pragma unroll
        for (int j = 0; j < 2; j++) {
            int n = nw + j * 16 + lr;
#pragma unroll
            for (int e = 0; e < 8; e++) {
                int c = mw + i * 16 + half * 8 + e;
                dst[(size_t)c * NSP + n] = f2bf(acc[i][j][e] + bias[c]);
            }
        }
    }
}

// ---------------- GEMM 2+3 fused: per-(b,h) attention core ----------------
// Phase 1: M = K_h * V_h^T  (64x64, registers), write M^T bf16 to LDS.
// Phase 2: Out = M^T * Q_h  streamed over n, stored transposed [b][n][c] bf16.
__global__ __launch_bounds__(256) void k_attn(const unsigned short* __restrict__ Kbuf,
                                              const unsigned short* __restrict__ Vbuf,
                                              const unsigned short* __restrict__ QT,
                                              unsigned short* __restrict__ AttnT) {
    __shared__ unsigned short MT[64 * 64];               // M^T, row-major [e][m], 8 KB
    int b = (int)blockIdx.x >> 3, h = (int)blockIdx.x & 7;
    int w = (int)(threadIdx.x >> 5);
    int lane = (int)(threadIdx.x & 31u);
    int half = lane >> 4, lr = lane & 15;

    const unsigned short* Kh = Kbuf + (size_t)b * CCH * NSP + (size_t)h * 64 * NSP;
    const unsigned short* Vh = Vbuf + (size_t)b * CCH * NSP + (size_t)h * 64 * NSP;

    // ---- phase 1: each wave owns tiles (mt, eb) and (mt, eb+1) of the 4x4 tile grid
    int mt = w & 3;
    int eb = (w >> 2) * 2;
    v8f acc0[2];
    acc0[0] = vzero8(); acc0[1] = vzero8();
    for (int n0 = 0; n0 < NSP; n0 += 32) {
        v16bf a  = load_a_frag(Kh, mt * 16, n0, NSP);
        v16bf b0 = load_b_frag(Vh, eb * 16,       n0, NSP);
        v16bf b1 = load_b_frag(Vh, (eb + 1) * 16, n0, NSP);
        acc0[0] = wmma_bf16(a, b0, acc0[0]);
        acc0[1] = wmma_bf16(a, b1, acc0[1]);
    }
#pragma unroll
    for (int t = 0; t < 2; t++)
#pragma unroll
    for (int e8 = 0; e8 < 8; e8++) {
        int m = mt * 16 + half * 8 + e8;                 // row of M
        int e = (eb + t) * 16 + lr;                      // col of M
        MT[e * 64 + m] = f2bf(acc0[t][e8]);
    }
    __syncthreads();

    // ---- phase 2: Out[e][n], wave covers e-block (w&1)*32, n-block (w>>1)*32 per 128-strip
    int e0 = (w & 1) * 32;
    int wn = w >> 1;
    const unsigned short* QTb = QT + (size_t)b * NSP * CCH + h * 64;   // row stride CCH, K = m
    unsigned short* dst = AttnT + (size_t)b * NSP * CCH;

    for (int nb = 0; nb < 8; nb++) {
        int nw = nb * 128 + wn * 32;
        v8f acc[2][2];
        acc[0][0] = vzero8(); acc[0][1] = vzero8(); acc[1][0] = vzero8(); acc[1][1] = vzero8();
#pragma unroll
        for (int m0 = 0; m0 < 64; m0 += 32) {
            v16bf a0 = load_a_frag(MT, e0,      m0, 64);
            v16bf a1 = load_a_frag(MT, e0 + 16, m0, 64);
            v16bf b0 = load_b_frag(QTb, nw,      m0, CCH);
            v16bf b1 = load_b_frag(QTb, nw + 16, m0, CCH);
            acc[0][0] = wmma_bf16(a0, b0, acc[0][0]);
            acc[0][1] = wmma_bf16(a0, b1, acc[0][1]);
            acc[1][0] = wmma_bf16(a1, b0, acc[1][0]);
            acc[1][1] = wmma_bf16(a1, b1, acc[1][1]);
        }
#pragma unroll
        for (int i = 0; i < 2; i++)
#pragma unroll
        for (int j = 0; j < 2; j++) {
            int n  = nw + j * 16 + lr;
            int cb = h * 64 + e0 + i * 16 + half * 8;
            union { v4u v; unsigned int u[4]; } pk;
#pragma unroll
            for (int r = 0; r < 4; r++) {
                unsigned lo = f2bf(acc[i][j][2 * r]);
                unsigned hi = f2bf(acc[i][j][2 * r + 1]);
                pk.u[r] = lo | (hi << 16);
            }
            *(v4u*)(dst + (size_t)n * CCH + cb) = pk.v;
        }
    }
}

// ---------------- GEMM 4: Wo conv + bias + residual + BN partial stats ----------------
__global__ __launch_bounds__(256) void k_out(const unsigned short* __restrict__ WoBf,   // [512][512]
                                             const unsigned short* __restrict__ AttnT,  // [B][n][c]
                                             const float* __restrict__ bo,
                                             const float* __restrict__ X,               // [B][c][n] f32
                                             float* __restrict__ Y,                     // [B][c][n] f32
                                             float* __restrict__ gsum,
                                             float* __restrict__ gsq) {
    __shared__ float ssum[64];
    __shared__ float ssq[64];
    int nt = blockIdx.x, mt = blockIdx.y, b = blockIdx.z;
    int m0 = mt * 64, n0 = nt * 128;
    if (threadIdx.x < 64) { ssum[threadIdx.x] = 0.f; ssq[threadIdx.x] = 0.f; }
    __syncthreads();

    int w = (int)(threadIdx.x >> 5);
    int lane = (int)(threadIdx.x & 31u);
    int half = lane >> 4, lr = lane & 15;
    int mw = m0 + (w & 1) * 32;
    int nw = n0 + (w >> 1) * 32;
    const unsigned short* Bt = AttnT + (size_t)b * NSP * CCH;

    v8f acc[2][2];
    acc[0][0] = vzero8(); acc[0][1] = vzero8(); acc[1][0] = vzero8(); acc[1][1] = vzero8();
    for (int k0 = 0; k0 < CCH; k0 += 32) {
        v16bf a0 = load_a_frag(WoBf, mw,      k0, CCH);
        v16bf a1 = load_a_frag(WoBf, mw + 16, k0, CCH);
        v16bf b0 = load_b_frag(Bt, nw,      k0, CCH);
        v16bf b1 = load_b_frag(Bt, nw + 16, k0, CCH);
        acc[0][0] = wmma_bf16(a0, b0, acc[0][0]);
        acc[0][1] = wmma_bf16(a0, b1, acc[0][1]);
        acc[1][0] = wmma_bf16(a1, b0, acc[1][0]);
        acc[1][1] = wmma_bf16(a1, b1, acc[1][1]);
    }

    const float* Xb = X + (size_t)b * CCH * NSP;
    float* Yb = Y + (size_t)b * CCH * NSP;
#pragma unroll
    for (int i = 0; i < 2; i++)
#pragma unroll
    for (int e8 = 0; e8 < 8; e8++) {
        int c = mw + i * 16 + half * 8 + e8;
        float bias = bo[c];
        float lsum = 0.f, lsq = 0.f;
#pragma unroll
        for (int j = 0; j < 2; j++) {
            int n = nw + j * 16 + lr;
            float v = acc[i][j][e8] + bias + Xb[(size_t)c * NSP + n];
            Yb[(size_t)c * NSP + n] = v;
            lsum += v;
            lsq  += v * v;
        }
        atomicAdd(&ssum[c - m0], lsum);
        atomicAdd(&ssq[c - m0], lsq);
    }
    __syncthreads();
    if (threadIdx.x < 64) {
        atomicAdd(&gsum[m0 + threadIdx.x], ssum[threadIdx.x]);
        atomicAdd(&gsq[m0 + threadIdx.x],  ssq[threadIdx.x]);
    }
}

// ---------------- BatchNorm finalize + apply ----------------
__global__ __launch_bounds__(256) void k_bnstats(const float* __restrict__ gsum, const float* __restrict__ gsq,
                                                 const float* __restrict__ gamma, const float* __restrict__ beta,
                                                 float* __restrict__ scl, float* __restrict__ sft) {
    int c = blockIdx.x * 256 + threadIdx.x;
    if (c >= CCH) return;
    float inv = 1.0f / (float)(NB * NSP);
    float mean = gsum[c] * inv;
    float var  = gsq[c] * inv - mean * mean;
    float s = gamma[c] * rsqrtf(var + EPSBN);
    scl[c] = s;
    sft[c] = beta[c] - mean * s;
}

__global__ __launch_bounds__(256) void k_bnapply(const float* __restrict__ Y,
                                                 const float* __restrict__ scl, const float* __restrict__ sft,
                                                 float* __restrict__ out) {
    int idx = blockIdx.x * 256 + threadIdx.x;            // 8*512*1024
    int c = (idx >> 10) & (CCH - 1);
    out[idx] = Y[idx] * scl[c] + sft[c];
}

// ---------------- launch ----------------
extern "C" void kernel_launch(void* const* d_in, const int* in_sizes, int n_in,
                              void* d_out, int out_size, void* d_ws, size_t ws_size,
                              hipStream_t stream) {
    const float* X     = (const float*)d_in[0];
    const float* Wq    = (const float*)d_in[1];
    const float* bq    = (const float*)d_in[2];
    const float* Wk    = (const float*)d_in[3];
    const float* bk    = (const float*)d_in[4];
    const float* Wv    = (const float*)d_in[5];
    const float* bv    = (const float*)d_in[6];
    const float* Wo    = (const float*)d_in[7];
    const float* bo    = (const float*)d_in[8];
    const float* gamma = (const float*)d_in[9];
    const float* beta  = (const float*)d_in[10];

    char* ws = (char*)d_ws;
    const size_t szW   = (size_t)4 * CCH * CCH * 2;          //  2 MB  bf16 weights (q,k,v,o)
    const size_t szBF  = (size_t)NB * NSP * CCH * 2;         //  8 MB  per bf16 tensor
    const size_t szY   = (size_t)NB * CCH * NSP * 4;         // 16 MB  f32 pre-BN

    unsigned short* Wbf = (unsigned short*)(ws);
    unsigned short* XT  = (unsigned short*)(ws + szW);
    unsigned short* Kb  = (unsigned short*)(ws + szW + szBF);
    unsigned short* Vb  = (unsigned short*)(ws + szW + 2 * szBF);
    unsigned short* QTp = (unsigned short*)(ws + szW + 3 * szBF);
    unsigned short* AT  = (unsigned short*)(ws + szW + 4 * szBF);
    float*          Yb  = (float*)(ws + szW + 5 * szBF);
    float*          gsum = (float*)(ws + szW + 5 * szBF + szY);
    float*          gsq  = gsum + 512;
    float*          scl  = gsum + 1024;
    float*          sft  = gsum + 1536;

    k_cvt_w<<<4096, 256, 0, stream>>>(Wq, Wk, Wv, Wo, Wbf);
    k_cvt_x<<<16384, 256, 0, stream>>>(X, XT);
    k_zero<<<8, 256, 0, stream>>>(gsum, 2048);
    k_gemm_qkv<<<dim3(8, 8, 24), 256, 0, stream>>>(Wbf, XT, bq, bk, bv, Kb, QTp, Vb);
    k_attn<<<64, 256, 0, stream>>>(Kb, Vb, QTp, AT);
    k_out<<<dim3(8, 8, 8), 256, 0, stream>>>(Wbf + (size_t)3 * CCH * CCH, AT, bo, X, Yb, gsum, gsq);
    k_bnstats<<<2, 256, 0, stream>>>(gsum, gsq, gamma, beta, scl, sft);
    k_bnapply<<<16384, 256, 0, stream>>>(Yb, scl, sft, (float*)d_out);
}
